// MultiHeadSelfAttention_59030030516776
// MI455X (gfx1250) — matrix operand order repaired and checked by reading, not compiled, not run on hardware
//
#include <hip/hip_runtime.h>
#include <hip/hip_bf16.h>

// MI455X / gfx1250 multi-head self-attention, bf16 WMMA path.
// B=2 S=2048 E=1024 H=16 D=64. ~69 GFLOP vs ~100MB HBM traffic -> compute
// bound; flash-style attention keeps the 32x16MB score matrices out of HBM.
// Round 4: native __bf16 conversions, k-pair-packed B staging (ds_store_b32),
// hoisted epilogue base pointers (constant-offset stores), exp2-domain
// softmax. Async global->LDS double-buffered K/V staging retained.

typedef unsigned short u16;
typedef __attribute__((ext_vector_type(16))) __bf16 v16bf;
typedef __attribute__((ext_vector_type(2)))  __bf16 v2bf;
typedef __attribute__((ext_vector_type(8)))  float  v8f;
typedef __attribute__((ext_vector_type(4)))  int    vint4;

static constexpr int Bb = 2, Ss = 2048, Ee = 1024, Hh = 16, Dd = 64;
static constexpr int Mm = Bb * Ss;        // 4096
static constexpr int N_QKV = 3 * Ee;      // 3072

// ---------- conversion helpers (native bf16 casts) ----------

__device__ __forceinline__ unsigned bfpack2(float a, float b) {
  v2bf v;
  v[0] = (__bf16)a;
  v[1] = (__bf16)b;
  return __builtin_bit_cast(unsigned, v);
}

__device__ __forceinline__ u16 f2bf(float f) {
  __bf16 h = (__bf16)f;
  return __builtin_bit_cast(u16, h);
}

__device__ __forceinline__ uint4 pack8(const float* __restrict__ f) {
  uint4 r;
  r.x = bfpack2(f[0], f[1]);
  r.y = bfpack2(f[2], f[3]);
  r.z = bfpack2(f[4], f[5]);
  r.w = bfpack2(f[6], f[7]);
  return r;
}

// ---------- async global->LDS (CDNA5, ASYNCcnt-tracked) ----------

#if __has_builtin(__builtin_amdgcn_global_load_async_to_lds_b128)
#define HAVE_ASYNC_LDS 1
#else
#define HAVE_ASYNC_LDS 0
#endif

__device__ __forceinline__ void async_copy16(void* lds_dst, const void* gsrc) {
#if HAVE_ASYNC_LDS
  __builtin_amdgcn_global_load_async_to_lds_b128(
      (__attribute__((address_space(1))) vint4*)gsrc,
      (__attribute__((address_space(3))) vint4*)lds_dst,
      0, 0);
#else
  *(uint4*)lds_dst = *(const uint4*)gsrc;
#endif
}

__device__ __forceinline__ void wait_async0() {
#if HAVE_ASYNC_LDS
#if __has_builtin(__builtin_amdgcn_s_wait_asynccnt)
  __builtin_amdgcn_s_wait_asynccnt(0);
#else
  asm volatile("s_wait_asynccnt 0x0" ::: "memory");
#endif
#endif
}

// ---------- fragment helpers ----------

struct BF16FragBits { uint4 lo, hi; };

__device__ __forceinline__ v16bf make_frag(const void* p0, const void* p1) {
  BF16FragBits t;
  t.lo = *(const uint4*)p0;
  t.hi = *(const uint4*)p1;
  return __builtin_bit_cast(v16bf, t);
}

__device__ __forceinline__ v8f v8zero() {
  v8f z;
#pragma unroll
  for (int i = 0; i < 8; ++i) z[i] = 0.0f;
  return z;
}

__device__ __forceinline__ v8f wmma_bf16(v16bf a, v16bf b, v8f c) {
  // D = A(16x32 bf16) x B(32x16 bf16) + C(16x16 f32)
  return __builtin_amdgcn_wmma_f32_16x16x32_bf16(
      false, a, false, b, (short)0, c, false, false);
}

// ---------- kernel 1: QKV projection ----------
// qkv = x @ Wqkv + bqkv ; scattered to q[b][h][s][d], k[b][h][s][d],
// v stored TRANSPOSED as vt[b][h][d][s] (so attention's P@V B-fragments are
// contiguous LDS reads).

__global__ __launch_bounds__(256)
void qkv_gemm_kernel(const float* __restrict__ x,     // [4096][1024] f32
                     const float* __restrict__ Wqkv,  // [1024][3072] f32
                     const float* __restrict__ bqkv,  // [3072]
                     u16* __restrict__ qd,            // [B][H][S][D] bf16
                     u16* __restrict__ kd,            // [B][H][S][D] bf16
                     u16* __restrict__ vtd)           // [B][H][D][S] bf16
{
  __shared__ __align__(16) u16 Asl[128][32];  // 8KB, row-major [m][k]
  __shared__ __align__(16) u16 Bsl[128][32];  // 8KB, transposed [n][k]
  unsigned* Bsl32 = (unsigned*)&Bsl[0][0];    // [n][k-pair] view

  const int m0 = blockIdx.y * 128;
  const int n0 = blockIdx.x * 128;
  const int t = threadIdx.x;
  const int lane = t & 31, wave = t >> 5;
  const int wm = wave >> 2, wn = wave & 3;        // 2x4 wave grid
  const int half = lane >> 4, ln = lane & 15;

  v8f acc[4][2];
#pragma unroll
  for (int mt = 0; mt < 4; ++mt)
#pragma unroll
    for (int nt = 0; nt < 2; ++nt) acc[mt][nt] = v8zero();

  const int ar = t >> 1, ac = (t & 1) * 16;       // A-tile stage coords
  const int bk2 = (t >> 4) << 1, bn8 = (t & 15) * 8;  // B-tile: 2 k-rows, 8 n

  for (int k0 = 0; k0 < Ee; k0 += 32) {
    // stage A tile (f32 -> bf16)
    const float* ap = x + (size_t)(m0 + ar) * Ee + k0 + ac;
    __builtin_prefetch(ap + 32, 0, 1);
    *(uint4*)&Asl[ar][ac]     = pack8(ap);
    *(uint4*)&Asl[ar][ac + 8] = pack8(ap + 8);
    // stage B tile transposed, k-pair packed (f32 -> bf16, ds_store_b32)
    const float* b0p = Wqkv + (size_t)(k0 + bk2) * N_QKV + n0 + bn8;
    const float* b1p = b0p + N_QKV;
    __builtin_prefetch(b0p + (size_t)32 * N_QKV, 0, 1);
#pragma unroll
    for (int i = 0; i < 8; ++i)
      Bsl32[(bn8 + i) * 16 + (bk2 >> 1)] = bfpack2(b0p[i], b1p[i]);
    __syncthreads();

    const int kb = half * 8;
    const int kh = half * 16;
    v16bf bfrag[2];
#pragma unroll
    for (int nt = 0; nt < 2; ++nt) {
      const int col = wn * 32 + nt * 16 + ln;
      bfrag[nt] = make_frag(&Bsl[col][kb], &Bsl[col][16 + kb]);
    }
#pragma unroll
    for (int mt = 0; mt < 4; ++mt) {
      const int row = wm * 64 + mt * 16 + ln;
      v16bf afrag = make_frag(&Asl[row][kb], &Asl[row][16 + kb]);
#pragma unroll
      for (int nt = 0; nt < 2; ++nt)
        acc[mt][nt] = wmma_bf16(afrag, bfrag[nt], acc[mt][nt]);
    }
    __syncthreads();
  }

  // epilogue: bias + scatter into q/k/vt (bf16), base pointers hoisted so the
  // unrolled stores use constant instruction offsets.
  const int mbase = m0 + wm * 64 + 8 * half;   // + mt*16 + v
  const int bIdx = mbase >> 11;                // constant over the tile
  const int sbase = mbase & 2047;
#pragma unroll
  for (int nt = 0; nt < 2; ++nt) {
    const int c = n0 + wn * 32 + nt * 16 + ln;
    const float bias = bqkv[c];
    const int three = c >> 10, rem = c & 1023;
    const int h = rem >> 6, d = rem & 63;
    if (three == 2) {
      u16* p = vtd + ((size_t)(bIdx * Hh + h) * Dd + d) * Ss + sbase;
#pragma unroll
      for (int mt = 0; mt < 4; ++mt)
#pragma unroll
        for (int v = 0; v < 8; ++v)
          p[mt * 16 + v] = f2bf(acc[mt][nt][v] + bias);
    } else {
      u16* p = (three ? kd : qd) +
               ((size_t)(bIdx * Hh + h) * Ss + sbase) * Dd + d;
#pragma unroll
      for (int mt = 0; mt < 4; ++mt)
#pragma unroll
        for (int v = 0; v < 8; ++v)
          p[(mt * 16 + v) * Dd] = f2bf(acc[mt][nt][v] + bias);
    }
  }
}

// ---------- kernel 2: flash attention ----------
// grid.y = b*H (32), grid.x = q-tile (128 rows). 8 waves, each wave owns a
// 16-row Q strip with online softmax state in registers. K/V tiles stream
// HBM->LDS via async-to-LDS with a ping-pong double buffer: the DMA of tile
// i+1 overlaps the WMMA/softmax work on tile i. Softmax runs in exp2 domain.

__global__ __launch_bounds__(256)
void attn_kernel(const u16* __restrict__ qd,    // [B][H][S][D] bf16
                 const u16* __restrict__ kd,    // [B][H][S][D] bf16
                 const u16* __restrict__ vtd,   // [B][H][D][S] bf16
                 const int* __restrict__ mask,  // [B][S] i32
                 u16* __restrict__ attn)        // [B*S][E] bf16
{
  constexpr int KT = 64;  // key-tile
  __shared__ __align__(16) u16 Ksl[2][KT][Dd];     // [buf][key][d] 16KB
  __shared__ __align__(16) u16 Vsl[2][Dd][KT];     // [buf][d][key] 16KB
  __shared__ __align__(16) u16 Psl[8][16][KT];     // per-wave P    16KB
  __shared__ float mb[2][KT];

  const int bh = blockIdx.y;            // b*H + h
  const int q0 = blockIdx.x * 128;
  const int batch = bh >> 4, h = bh & 15;
  const int t = threadIdx.x, lane = t & 31, wave = t >> 5;
  const int half = lane >> 4, ln = lane & 15;
  const int kb = half * 8, kh = half * 16;

  const u16* qbase = qd + (size_t)bh * Ss * Dd;
  const u16* kbase = kd + (size_t)bh * Ss * Dd;
  const u16* vbase = vtd + (size_t)bh * Dd * Ss;

  // Q strip A-fragments (d 0..31 and 32..63), resident for whole kernel
  const int qrow = q0 + wave * 16 + ln;
  const u16* qp = qbase + (size_t)qrow * Dd;
  const v16bf aq0 = make_frag(qp + kb, qp + 16 + kb);
  const v16bf aq1 = make_frag(qp + 32 + kb, qp + 48 + kb);

  float mrow[8], lrow[8];
  v8f O[4];
#pragma unroll
  for (int v = 0; v < 8; ++v) { mrow[v] = -3.0e30f; lrow[v] = 0.0f; }
#pragma unroll
  for (int nt = 0; nt < 4; ++nt) O[nt] = v8zero();

  // exp2-domain softmax: scale = log2(e)/sqrt(D); exp2f -> v_exp_f32
  const float scale2 = 1.4426950408889634f * 0.125f;

  // cooperative stage of one K/V tile into buffer bufi (async, no wait)
  auto stage_kv = [&](int kt0, int bufi) {
    const int r = t >> 2, cb = (t & 3) * 16;       // r: key for K / d for V
    const u16* ksrc = kbase + (size_t)(kt0 + r) * Dd + cb;
    const u16* vsrc = vbase + (size_t)r * Ss + kt0 + cb;
    async_copy16(&Ksl[bufi][r][cb],     ksrc);
    async_copy16(&Ksl[bufi][r][cb + 8], ksrc + 8);
    async_copy16(&Vsl[bufi][r][cb],     vsrc);
    async_copy16(&Vsl[bufi][r][cb + 8], vsrc + 8);
    if (t < KT) mb[bufi][t] = mask[batch * Ss + kt0 + t] ? 0.0f : -3.0e30f;
  };

  stage_kv(0, 0);
  int buf = 0;

  for (int kt0 = 0; kt0 < Ss; kt0 += KT) {
    wait_async0();       // this wave's async loads into `buf` are complete
    __syncthreads();     // every wave's loads are complete

    if (kt0 + KT < Ss) stage_kv(kt0 + KT, buf ^ 1);  // overlap next DMA

    // scores: Q strip (16x64) @ K^T (64x64) -> 4 x (16x16) f32 fragments
    v8f sc[4];
    float mbl[4];
#pragma unroll
    for (int nt = 0; nt < 4; ++nt) {
      const int key = nt * 16 + ln;
      v16bf b0 = make_frag(&Ksl[buf][key][kb], &Ksl[buf][key][16 + kb]);
      v16bf b1 = make_frag(&Ksl[buf][key][32 + kb], &Ksl[buf][key][48 + kb]);
      v8f z = v8zero();
      z = wmma_bf16(aq0, b0, z);
      z = wmma_bf16(aq1, b1, z);
      sc[nt] = z;
      mbl[nt] = mb[buf][key];
    }

    // online softmax update (rows live in one 16-lane half: xor 1..8 reduce)
#pragma unroll
    for (int v = 0; v < 8; ++v) {
      float mx = -3.0e30f;
#pragma unroll
      for (int nt = 0; nt < 4; ++nt) {
        float s = sc[nt][v] * scale2 + mbl[nt];
        sc[nt][v] = s;
        mx = fmaxf(mx, s);
      }
#pragma unroll
      for (int off = 1; off < 16; off <<= 1)
        mx = fmaxf(mx, __shfl_xor(mx, off, 32));
      const float newm = fmaxf(mrow[v], mx);
      const float corr = exp2f(mrow[v] - newm);
      float rs = 0.0f;
#pragma unroll
      for (int nt = 0; nt < 4; ++nt) {
        float p = exp2f(sc[nt][v] - newm);
        sc[nt][v] = p;
        rs += p;
      }
#pragma unroll
      for (int off = 1; off < 16; off <<= 1) rs += __shfl_xor(rs, off, 32);
      lrow[v] = lrow[v] * corr + rs;
      mrow[v] = newm;
#pragma unroll
      for (int nt = 0; nt < 4; ++nt) O[nt][v] *= corr;
    }

    // re-layout P: C-fragment -> per-wave LDS -> A-fragment (LDS ops are
    // in-order within a wave; Psl[wave] is wave-private)
#pragma unroll
    for (int v = 0; v < 8; ++v) {
      const int r = v + 8 * half;
#pragma unroll
      for (int nt = 0; nt < 4; ++nt)
        Psl[wave][r][nt * 16 + ln] = f2bf(sc[nt][v]);
    }
    v16bf p0 = make_frag(&Psl[wave][ln][kb], &Psl[wave][ln][16 + kb]);
    v16bf p1 = make_frag(&Psl[wave][ln][32 + kb], &Psl[wave][ln][48 + kb]);

    // O += P (16x64) @ V (64x64)
#pragma unroll
    for (int nt = 0; nt < 4; ++nt) {
      const int dg = nt * 16 + ln;
      v16bf b0 = make_frag(&Vsl[buf][dg][kb], &Vsl[buf][dg][16 + kb]);
      v16bf b1 = make_frag(&Vsl[buf][dg][32 + kb], &Vsl[buf][dg][48 + kb]);
      O[nt] = wmma_bf16(p0, b0, O[nt]);
      O[nt] = wmma_bf16(p1, b1, O[nt]);
    }
    __syncthreads();   // all waves done reading `buf` before it is re-staged
    buf ^= 1;
  }

  // normalize and write attn[b*S+s][h*64+d] as bf16 (base pointer hoisted,
  // unrolled stores use constant instruction offsets)
  {
    const int srow0 = q0 + wave * 16 + 8 * half;   // + v
    u16* p = attn + ((size_t)batch * Ss + srow0) * Ee + h * Dd + ln;
#pragma unroll
    for (int v = 0; v < 8; ++v) {
      const float inv = 1.0f / fmaxf(lrow[v], 1e-20f);
#pragma unroll
      for (int nt = 0; nt < 4; ++nt)
        p[v * Ee + nt * 16] = f2bf(O[nt][v] * inv);
    }
  }
}

// ---------- kernel 3: output projection ----------
// out = attn(bf16) @ Wout(f32->bf16) + bout, f32 result to d_out.
// A tile is already bf16 in memory -> async global->LDS copy.

__global__ __launch_bounds__(256)
void out_gemm_kernel(const u16* __restrict__ attn,   // [4096][1024] bf16
                     const float* __restrict__ Wout, // [1024][1024] f32
                     const float* __restrict__ bout, // [1024]
                     float* __restrict__ out)        // [4096][1024] f32
{
  __shared__ __align__(16) u16 Asl[128][32];
  __shared__ __align__(16) u16 Bsl[128][32];  // transposed [n][k]
  unsigned* Bsl32 = (unsigned*)&Bsl[0][0];    // [n][k-pair] view

  const int m0 = blockIdx.y * 128;
  const int n0 = blockIdx.x * 128;
  const int t = threadIdx.x;
  const int lane = t & 31, wave = t >> 5;
  const int wm = wave >> 2, wn = wave & 3;
  const int half = lane >> 4, ln = lane & 15;

  v8f acc[4][2];
#pragma unroll
  for (int mt = 0; mt < 4; ++mt)
#pragma unroll
    for (int nt = 0; nt < 2; ++nt) acc[mt][nt] = v8zero();

  const int ar = t >> 1, ac = (t & 1) * 16;
  const int bk2 = (t >> 4) << 1, bn8 = (t & 15) * 8;

  for (int k0 = 0; k0 < Ee; k0 += 32) {
    // A tile: bf16 already -> async straight into LDS
    const u16* ap = attn + (size_t)(m0 + ar) * Ee + k0 + ac;
    async_copy16(&Asl[ar][ac], ap);
    async_copy16(&Asl[ar][ac + 8], ap + 8);
    // B tile: f32 -> bf16 transposed, k-pair packed
    const float* b0p = Wout + (size_t)(k0 + bk2) * Ee + n0 + bn8;
    const float* b1p = b0p + Ee;
    __builtin_prefetch(b0p + (size_t)32 * Ee, 0, 1);
#pragma unroll
    for (int i = 0; i < 8; ++i)
      Bsl32[(bn8 + i) * 16 + (bk2 >> 1)] = bfpack2(b0p[i], b1p[i]);
    wait_async0();
    __syncthreads();

    const int kb = half * 8;
    const int kh = half * 16;
    v16bf bfrag[2];
#pragma unroll
    for (int nt = 0; nt < 2; ++nt) {
      const int col = wn * 32 + nt * 16 + ln;
      bfrag[nt] = make_frag(&Bsl[col][kb], &Bsl[col][16 + kb]);
    }
#pragma unroll
    for (int mt = 0; mt < 4; ++mt) {
      const int row = wm * 64 + mt * 16 + ln;
      v16bf afrag = make_frag(&Asl[row][kb], &Asl[row][16 + kb]);
#pragma unroll
      for (int nt = 0; nt < 2; ++nt)
        acc[mt][nt] = wmma_bf16(afrag, bfrag[nt], acc[mt][nt]);
    }
    __syncthreads();
  }

  // epilogue: base pointer hoisted per nt; stores use constant offsets
#pragma unroll
  for (int nt = 0; nt < 2; ++nt) {
    const int c = n0 + wn * 32 + nt * 16 + ln;
    const float bias = bout[c];
    float* p = out + (size_t)(m0 + wm * 64 + 8 * half) * Ee + c;
#pragma unroll
    for (int mt = 0; mt < 4; ++mt)
#pragma unroll
      for (int v = 0; v < 8; ++v)
        p[(size_t)(mt * 16 + v) * Ee] = acc[mt][nt][v] + bias;
  }
}

// ---------- launcher ----------

extern "C" void kernel_launch(void* const* d_in, const int* in_sizes, int n_in,
                              void* d_out, int out_size, void* d_ws, size_t ws_size,
                              hipStream_t stream) {
  const float* x    = (const float*)d_in[0];   // [B][S][E]
  const int*   mask = (const int*)d_in[1];     // [B][S]
  const float* Wqkv = (const float*)d_in[2];   // [E][3E]
  const float* bqkv = (const float*)d_in[3];   // [3E]
  const float* Wout = (const float*)d_in[4];   // [E][E]
  const float* bout = (const float*)d_in[5];   // [E]
  float* out = (float*)d_out;                  // [B][S][E]

  // workspace: q, k (b,h,s,d) + vt (b,h,d,s) + attn (b*s, e), all bf16 = 32MB
  const size_t qsz = (size_t)Bb * Hh * Ss * Dd;   // elements
  u16* qd   = (u16*)d_ws;
  u16* kd   = qd + qsz;
  u16* vtd  = kd + qsz;
  u16* attn = vtd + qsz;

  qkv_gemm_kernel<<<dim3(N_QKV / 128, Mm / 128), 256, 0, stream>>>(
      x, Wqkv, bqkv, qd, kd, vtd);
  attn_kernel<<<dim3(Ss / 128, Bb * Hh), 256, 0, stream>>>(
      qd, kd, vtd, mask, attn);
  out_gemm_kernel<<<dim3(Ee / 128, Mm / 128), 256, 0, stream>>>(
      attn, Wout, bout, out);
}
